// Critic_11544872091972
// MI455X (gfx1250) — matrix operand order repaired and checked by reading, not compiled, not run on hardware
//
#include <hip/hip_runtime.h>

#define NSEQ 4096
#define TT   256
#define OBS  32
#define ACT  8
#define HH   64
#define DIN  104      // OBS+ACT+H
#define GATES 256     // 4*H
#define MT   16       // batch rows per block
#define ASTRIDE 136   // padded K stride (halves); 272B = 16B-aligned rows

typedef __attribute__((ext_vector_type(16))) _Float16 v16h;
typedef __attribute__((ext_vector_type(8)))  float    v8f;
typedef __attribute__((ext_vector_type(4)))  unsigned int u32x4;
typedef __attribute__((ext_vector_type(2)))  unsigned int u32x2;
typedef __attribute__((ext_vector_type(4)))  float    f32x4;
typedef __attribute__((ext_vector_type(2)))  float    f32x2;

union V16H { v16h v; _Float16 e[16]; u32x4 q[2]; };
union V8F  { v8f v; float f[8]; };
union HPK  { unsigned int u; _Float16 h[2]; };

// CDNA5 hardware tanh (TRANS unit). Builtin if present, else raw ISA.
__device__ __forceinline__ float htanh_(float x) {
#if __has_builtin(__builtin_amdgcn_tanhf)
    return __builtin_amdgcn_tanhf(x);
#else
    float r;
    asm volatile("v_tanh_f32 %0, %1" : "=v"(r) : "v"(x));
    return r;
#endif
}
// sigmoid(x) = 0.5 + 0.5*tanh(0.5*x)  -> 1 trans + 2 FMA-class ops
__device__ __forceinline__ float sigmoid_(float x) {
    return __builtin_fmaf(0.5f, htanh_(0.5f * x), 0.5f);
}

__global__ __launch_bounds__(256) void lstm_wmma_kernel(
    const float* __restrict__ obss,     // [N, T, OBS]
    const float* __restrict__ actions,  // [N, T, ACT]
    const float* __restrict__ Wl,       // [DIN, 4H]
    const float* __restrict__ bl,       // [4H]
    const float* __restrict__ Wd,       // [H, 1]
    const float* __restrict__ bd,       // [1]
    float* __restrict__ out)            // [T, N, 1]
{
    __shared__ __align__(16) _Float16 A[MT * ASTRIDE];   // x_t || h_{t-1}, f16, K-padded
    __shared__ __align__(16) float    G[MT * GATES];     // gate pre-activations

    const int tid  = threadIdx.x;
    const int lane = tid & 31;
    const int wave = tid >> 5;          // 0..7, owns gate cols [32w, 32w+32)
    const int row0 = blockIdx.x * MT;

    // ---------------- B fragments: weights -> registers, held for all T ----
    // B layout used here (16-bit, 32x16, contiguous K, same as A): lanes 0-15 hold K=kb..kb+15 for col N=lane,
    // lanes 16-31 hold K=kb+16..kb+31 for col N=lane-16.
    const int nlo   = lane & 15;
    const int khalf = (lane < 16) ? 0 : 16;
    V16H B[2][4];
    for (int nt = 0; nt < 2; ++nt) {
        const int n = (wave * 2 + nt) * 16 + nlo;
        for (int ks = 0; ks < 4; ++ks) {
            const int kb = ks * 32 + khalf;
            #pragma unroll
            for (int j = 0; j < 16; ++j) {
                const int k = kb + j;
                B[nt][ks].e[j] = (k < DIN) ? (_Float16)Wl[k * GATES + n] : (_Float16)0.0f;
            }
        }
    }
    const float bias0 = bl[wave * 32 + nlo];
    const float bias1 = bl[wave * 32 + 16 + nlo];

    // A-fragment per-lane addressing (contiguous K = 16*(lane/16) + s, same as B, 16-bit 16x32)
    const int am  = lane & 15;                 // row M
    const int akb = (lane < 16) ? 0 : 16;      // K sub-offset

    // elementwise mapping: thread -> (row, 4 consecutive k)
    const int erow = tid >> 4;                 // 0..15
    const int ek0  = (tid & 15) * 4;           // 0..60
    float c0 = 0.f, c1 = 0.f, c2 = 0.f, c3 = 0.f;   // cell state in registers
    const float wd0 = Wd[ek0], wd1 = Wd[ek0 + 1], wd2 = Wd[ek0 + 2], wd3 = Wd[ek0 + 3];
    const float bdec = bd[0];

    // per-thread source pointers for x staging
    const int xo = (tid & 15) * 2;                       // obs offset (2 floats)
    const float* obs_src = &obss[(row0 + erow) * TT * OBS + xo];
    // action staging: threads 128-255 duplicate loads of threads 0-127
    // (coalesced, L2-resident) so the load needs no EXEC branch and can be
    // hoisted into the WMMA phase; only the LDS store is predicated.
    const int arow = (tid >> 3) & 15, acol = tid & 7;
    const float* act_src = &actions[(row0 + arow) * TT * ACT + acol];

    // ---------------- init: zero A (covers h0=0 and K padding) -------------
    unsigned int* A32 = (unsigned int*)A;
    for (int i = tid; i < (MT * ASTRIDE) / 2; i += 256) A32[i] = 0u;
    __syncthreads();

    // stage x_0
    {
        const f32x2 v = *(const f32x2*)(obs_src + 0 * OBS);
        HPK p; p.h[0] = (_Float16)v.x; p.h[1] = (_Float16)v.y;
        *(unsigned int*)&A[erow * ASTRIDE + xo] = p.u;
        if (tid < 128) {
            A[arow * ASTRIDE + OBS + acol] = (_Float16)act_src[0 * ACT];
        }
    }

    // ---------------- main recurrence ------------------------------------
    for (int t = 0; t < TT; ++t) {
        // branchless clamped staging indices (last-iteration store is dead)
        const int tnx = (t + 1 < TT) ? (t + 1) : t;
        const int tpf = (t + 2 < TT) ? (t + 2) : t;

        __syncthreads();   // A = [x_t || h_{t-1}] fully staged

        // issue x_{t+1} global loads FIRST: latency hidden under WMMA phase
        const f32x2 xnext = *(const f32x2*)(obs_src + tnx * OBS);
        const float anext = act_src[tnx * ACT];
        __builtin_prefetch(obs_src + tpf * OBS, 0, 0);
        __builtin_prefetch(act_src + tpf * ACT, 0, 0);

        // load A fragments from LDS (two 16B chunks each, 16B aligned)
        V16H Af[4];
        #pragma unroll
        for (int ks = 0; ks < 4; ++ks) {
            const int base = am * ASTRIDE + ks * 32 + akb;
            Af[ks].q[0] = *(const u32x4*)&A[base];
            Af[ks].q[1] = *(const u32x4*)&A[base + 8];
        }

        V8F acc0, acc1;
        #pragma unroll
        for (int r = 0; r < 8; ++r) { acc0.f[r] = bias0; acc1.f[r] = bias1; }

        #pragma unroll
        for (int ks = 0; ks < 4; ++ks) {
            acc0.v = __builtin_amdgcn_wmma_f32_16x16x32_f16(
                false, Af[ks].v, false, B[0][ks].v, (short)0, acc0.v, false, false);
            acc1.v = __builtin_amdgcn_wmma_f32_16x16x32_f16(
                false, Af[ks].v, false, B[1][ks].v, (short)0, acc1.v, false, false);
        }

        // scatter C/D fragment to gates LDS (ISA C layout: VGPR r -> row r / r+8)
        const int gcol = wave * 32 + nlo;
        #pragma unroll
        for (int r = 0; r < 8; ++r) {
            const int gr = (lane < 16) ? r : (r + 8);
            G[gr * GATES + gcol]      = acc0.f[r];
            G[gr * GATES + gcol + 16] = acc1.f[r];
        }

        __syncthreads();   // gates ready; all A reads retired

        // ------- elementwise LSTM cell + decode (gate order: i, j, f, o) ---
        const f32x4* grow = (const f32x4*)&G[erow * GATES];
        const int q = ek0 >> 2;
        const f32x4 gi = grow[q], gj = grow[16 + q], gf = grow[32 + q], go = grow[48 + q];

        c0 = c0 * sigmoid_(gf.x + 1.0f) + sigmoid_(gi.x) * htanh_(gj.x);
        c1 = c1 * sigmoid_(gf.y + 1.0f) + sigmoid_(gi.y) * htanh_(gj.y);
        c2 = c2 * sigmoid_(gf.z + 1.0f) + sigmoid_(gi.z) * htanh_(gj.z);
        c3 = c3 * sigmoid_(gf.w + 1.0f) + sigmoid_(gi.w) * htanh_(gj.w);
        const float h0 = htanh_(c0) * sigmoid_(go.x);
        const float h1 = htanh_(c1) * sigmoid_(go.y);
        const float h2 = htanh_(c2) * sigmoid_(go.z);
        const float h3 = htanh_(c3) * sigmoid_(go.w);

        // write h_t back into A staging (f16) for next step
        HPK pa; pa.h[0] = (_Float16)h0; pa.h[1] = (_Float16)h1;
        HPK pb; pb.h[0] = (_Float16)h2; pb.h[1] = (_Float16)h3;
        u32x2 hw; hw.x = pa.u; hw.y = pb.u;
        *(u32x2*)&A[erow * ASTRIDE + (OBS + ACT) + ek0] = hw;

        // decode: 16-lane tree reduction per row
        float part = h0 * wd0 + h1 * wd1 + h2 * wd2 + h3 * wd3;
        #pragma unroll
        for (int off = 8; off > 0; off >>= 1) part += __shfl_xor(part, off, 16);
        if ((tid & 15) == 0) out[t * NSEQ + row0 + erow] = part + bdec;

        // commit pre-loaded x_{t+1} into A staging
        HPK px; px.h[0] = (_Float16)xnext.x; px.h[1] = (_Float16)xnext.y;
        *(unsigned int*)&A[erow * ASTRIDE + xo] = px.u;
        if (tid < 128) {
            A[arow * ASTRIDE + OBS + acol] = (_Float16)anext;
        }
    }
}

extern "C" void kernel_launch(void* const* d_in, const int* in_sizes, int n_in,
                              void* d_out, int out_size, void* d_ws, size_t ws_size,
                              hipStream_t stream) {
    (void)in_sizes; (void)n_in; (void)out_size; (void)d_ws; (void)ws_size;
    const float* obss    = (const float*)d_in[0];
    const float* actions = (const float*)d_in[1];
    const float* Wl      = (const float*)d_in[2];
    const float* bl      = (const float*)d_in[3];
    const float* Wd      = (const float*)d_in[4];
    const float* bd      = (const float*)d_in[5];
    float* out = (float*)d_out;

    dim3 grid(NSEQ / MT);   // 256 blocks of 16 batch rows
    dim3 block(256);        // 8 waves (wave32)
    lstm_wmma_kernel<<<grid, block, 0, stream>>>(obss, actions, Wl, bl, Wd, bd, out);
}
